// AttentionModel_39848706572743
// MI455X (gfx1250) — matrix operand order repaired and checked by reading, not compiled, not run on hardware
//
#include <hip/hip_runtime.h>

typedef __attribute__((ext_vector_type(16))) _Float16 v16h;
typedef __attribute__((ext_vector_type(8)))  _Float16 v8h;
typedef __attribute__((ext_vector_type(8)))  float    v8f;

#define WMMA_F32_F16(A, B, C) \
    __builtin_amdgcn_wmma_f32_16x16x32_f16(false, (A), false, (B), (short)0, (C), false, false)

// Problem constants (match reference setup_inputs)
constexpr int Bsz = 2;
constexpr int S   = 2048;
constexpr int H   = 16;
constexpr int D   = 64;
constexpr float SCALE   = 8.0f;          // reference: scores / (1/sqrt(D)) == scores * 8
constexpr float DROP_P  = 0.1f;
constexpr float KEEP_SC = 1.0f / 0.9f;

__device__ __forceinline__ v16h load16h(const _Float16* p) {
    const v8h lo = *(const v8h*)p;
    const v8h hi = *(const v8h*)(p + 16);
    return __builtin_shufflevector(lo, hi, 0,1,2,3,4,5,6,7,8,9,10,11,12,13,14,15);
}

__global__ __launch_bounds__(128) void fa_fwd_kernel(
    const float* __restrict__ Q, const float* __restrict__ K,
    const float* __restrict__ V, const float* __restrict__ Msk,
    const float* __restrict__ U, float* __restrict__ O)
{
    const int tid  = threadIdx.x;
    const int wave = tid >> 5;
    const int lane = tid & 31;
    const int n16  = lane & 15;        // tile column (C-domain) / q-row (row-domain)
    const int hs   = lane >> 4;        // half-wave select

    constexpr int QBLK = 64;           // rows per block (4 waves x 16)
    const int qblocks = S / QBLK;
    const int bh   = blockIdx.x / qblocks;
    const int qblk = blockIdx.x % qblocks;
    const int b    = bh / H;
    const int h    = bh % H;
    const int q0   = qblk * QBLK + wave * 16;

    const size_t rowStride = (size_t)H * D;  // [B,S,H,D]: stride between s indices
    const float* Qb = (const float*)__builtin_assume_aligned(Q, 16) + ((size_t)b * S) * rowStride + (size_t)h * D;
    const float* Kb = (const float*)__builtin_assume_aligned(K, 16) + ((size_t)b * S) * rowStride + (size_t)h * D;
    const float* Vb = (const float*)__builtin_assume_aligned(V, 16) + ((size_t)b * S) * rowStride + (size_t)h * D;
    const float* Mb = Msk + (((size_t)b * H + h) * S) * S;
    const float* Ub = U   + (((size_t)b * H + h) * S) * S;

    // Cooperative f16 staging (shared by all 4 waves):
    //  ldsK : keys row-major [k][d], rows padded to 72 halves (144B: 16B-aligned,
    //         conflict-free b128 fragment reads)
    //  ldsVt: values transposed [d][k], rows padded to 40 halves (80B: aligned,
    //         conflict-free b128 fragment reads)
    //  ldsS : per-wave raw-score tile (f32, C-layout -> row-major transpose),
    //         rows padded to 36 floats (144B: aligned + conflict-free both ways)
    __shared__ __align__(16) _Float16 ldsK [32][72];
    __shared__ __align__(16) _Float16 ldsVt[64][40];
    __shared__ __align__(16) float    ldsS [4][16][36];

    // staging assignment: each thread owns key-row (tid&31), 16-dim group (tid>>5)
    const int skr = tid & 31;
    const int sg  = tid >> 5;

    // ---- Load Q A-fragments (16x32 f16 each, two d-chunks), fold sqrt(D) scale
    v16h qa[2];
    {
        const float* qrow = Qb + (size_t)(q0 + n16) * rowStride;
        #pragma unroll
        for (int dc = 0; dc < 2; ++dc) {
            #pragma unroll
            for (int i = 0; i < 16; ++i) {
                const int kloc = (i & 7) + ((i >> 3) << 4) + hs * 8;   // ISA A f16 16x32 layout
                qa[dc][i] = (_Float16)(SCALE * qrow[dc * 32 + kloc]);
            }
        }
    }

    // ---- Online-softmax state: one q-row per lane in the row-major domain
    float mrow_s = -3.0e38f;   // running max of row (q0 + n16)
    float lrow_s = 0.0f;       // running pre-dropout denominator of that row
    v8f acc[4];
    #pragma unroll
    for (int c = 0; c < 4; ++c) acc[c] = v8f{};

    // mask/dropout pointers for this lane's q-row (row-major, vectorizable)
    const size_t myRowOff = (size_t)(q0 + n16) * S;

    for (int kk = 0; kk < S; kk += 32) {
        // ============ cooperative staging of K tile (32x64) and V tile (32x64)
        __syncthreads();   // previous iteration's LDS reads complete
        {
            const float* krow = Kb + (size_t)(kk + skr) * rowStride + sg * 16;
            const float* vrow = Vb + (size_t)(kk + skr) * rowStride + sg * 16;
            _Float16* kdst = &ldsK[skr][sg * 16];
            #pragma unroll
            for (int j = 0; j < 16; ++j) kdst[j] = (_Float16)krow[j];
            #pragma unroll
            for (int j = 0; j < 16; ++j) ldsVt[sg * 16 + j][skr] = (_Float16)vrow[j];
        }
        __syncthreads();   // staged tiles visible

        // ============ scores: two 16x16 tiles over d=64 (2 WMMA K-chunks each)
        v8f s0 = v8f{}, s1 = v8f{};
        #pragma unroll
        for (int dc = 0; dc < 2; ++dc) {
            const v16h kb0 = load16h(&ldsK[n16]     [dc * 32 + hs * 8]);
            const v16h kb1 = load16h(&ldsK[16 + n16][dc * 32 + hs * 8]);
            s0 = WMMA_F32_F16(qa[dc], kb0, s0);
            s1 = WMMA_F32_F16(qa[dc], kb1, s1);
        }

        // ============ transpose raw scores through LDS (C-layout -> row-major)
        #pragma unroll
        for (int r = 0; r < 8; ++r) {
            ldsS[wave][r + hs * 8][n16]      = s0[r];
            ldsS[wave][r + hs * 8][16 + n16] = s1[r];
        }
        __syncthreads();   // cross-lane visibility of score tile

        // lane now owns q-row n16; it holds columns [hs*8, hs*8+8) and [16+hs*8, ...)
        const float* srow = &ldsS[wave][n16][0];
        const float* mp = Mb + myRowOff + kk + hs * 8;   // 2x 32B contiguous
        const float* up = Ub + myRowOff + kk + hs * 8;
        if (kk + 32 < S) {
            __builtin_prefetch(mp + 32, 0, 0);           // global_prefetch_b8
            __builtin_prefetch(up + 32, 0, 0);
        }

        float sv[16];
        #pragma unroll
        for (int i = 0; i < 8; ++i) {
            sv[i]     = srow[hs * 8 + i]      + mp[i];        // + attn_mask
            sv[8 + i] = srow[16 + hs * 8 + i] + mp[16 + i];
        }

        // ============ online softmax (local tree + one cross-half shuffle)
        float mx = sv[0];
        #pragma unroll
        for (int i = 1; i < 16; ++i) mx = fmaxf(mx, sv[i]);
        mx = fmaxf(mx, __shfl_xor(mx, 16, 32));              // combine the two half-rows

        const float mn   = fmaxf(mrow_s, mx);
        const float corr = __expf(mrow_s - mn);
        mrow_s = mn;

        float psum = 0.0f;
        float pk[16];
        #pragma unroll
        for (int i = 0; i < 16; ++i) {
            const float p = __expf(sv[i] - mn);
            psum += p;                                        // pre-dropout sum
            const float u = (i < 8) ? up[i] : up[16 + (i - 8)];
            pk[i] = p * ((u >= DROP_P) ? KEEP_SC : 0.0f);     // post-softmax dropout
        }
        psum += __shfl_xor(psum, 16, 32);
        lrow_s = lrow_s * corr + psum;

        // P A-fragment built directly in registers (already in A-layout!)
        v16h pa;
        #pragma unroll
        for (int i = 0; i < 16; ++i) pa[i] = (_Float16)pk[i];

        // broadcast per-row corr into the C-layout accumulator domain
        #pragma unroll
        for (int r = 0; r < 8; ++r) {
            const float cr = __shfl(corr, hs * 8 + r, 32);
            acc[0][r] *= cr; acc[1][r] *= cr;
            acc[2][r] *= cr; acc[3][r] *= cr;
        }

        // ============ O += P @ V  (4 output d-chunks of 16)
        #pragma unroll
        for (int c = 0; c < 4; ++c) {
            const v16h vb = load16h(&ldsVt[c * 16 + n16][hs * 8]);
            acc[c] = WMMA_F32_F16(pa, vb, acc[c]);
        }
    }

    // ============ epilogue: normalize by softmax denominator, write [B,H,S,D]
    const float invl = 1.0f / lrow_s;                         // per q-row (row-domain)
    #pragma unroll
    for (int r = 0; r < 8; ++r) {
        const float inv = __shfl(invl, hs * 8 + r, 32);       // to C-layout rows
        const int row = q0 + r + hs * 8;
        float* orow = O + (((size_t)bh) * S + row) * D;
        #pragma unroll
        for (int c = 0; c < 4; ++c)
            orow[c * 16 + n16] = acc[c][r] * inv;
    }
}

extern "C" void kernel_launch(void* const* d_in, const int* in_sizes, int n_in,
                              void* d_out, int out_size, void* d_ws, size_t ws_size,
                              hipStream_t stream) {
    (void)in_sizes; (void)n_in; (void)d_ws; (void)ws_size; (void)out_size;
    const float* q = (const float*)d_in[0];
    const float* k = (const float*)d_in[1];
    const float* v = (const float*)d_in[2];
    const float* m = (const float*)d_in[3];
    const float* u = (const float*)d_in[4];
    float* out = (float*)d_out;

    const int grid = Bsz * H * (S / 64);   // 1024 workgroups of 4 waves
    fa_fwd_kernel<<<grid, 128, 0, stream>>>(q, k, v, m, u, out);
}
